// Net_51513837748289
// MI455X (gfx1250) — matrix operand order repaired and checked by reading, not compiled, not run on hardware
//
#include <hip/hip_runtime.h>

// 2-layer LSTM (H=30, B=1024, T=2048, D_in=1) + sigmoid(relu(h1)@Wout^T)
// Persistent kernel: 64 blocks x 128 threads (4 waves). Each block owns a
// batch tile of 16; each wave owns 2 of the 8 gate N-tiles (gates padded
// 120->128) and keeps its f16 weight WMMA B-fragments in registers for the
// entire sequence. Per step/wave: 2 WMMAs (layer0) + 4 chained WMMAs (layer1)
// using V_WMMA_F32_16X16X32_F16 (K = 30 padded to 32).
//
// Round-2 changes:
//  - fast rcp (v_rcp_f32) for sigmoid/tanh instead of IEEE divide chains
//  - element-wise phase: fixed (q,j) per thread, one uniform branch, no
//    per-iteration exec-mask churn, no integer divide in the loop
//  - x staged in 32-step chunks (transposed in LDS), next chunk's global
//    loads issued 32 steps early into registers (latency hidden)

typedef __attribute__((ext_vector_type(16))) _Float16 v16h;
typedef __attribute__((ext_vector_type(8)))  _Float16 v8h;
typedef __attribute__((ext_vector_type(8)))  float    v8f;

#define HH   30      // hidden size
#define G4   120     // 4*H gate rows
#define TT   2048    // sequence length
#define BB   1024    // batch
#define CH   32      // x staging chunk (steps)

__device__ __forceinline__ float rcp_(float x) {
    return __builtin_amdgcn_rcpf(x);          // v_rcp_f32, no div fixup
}
__device__ __forceinline__ float sigf(float x) {
    return rcp_(1.0f + __expf(-x));
}
__device__ __forceinline__ float tanh_(float x) {
    const float e = __expf(-2.0f * x);
    return (1.0f - e) * rcp_(1.0f + e);
}

// Build a 32x16 (KxN) f16 B-fragment for WMMA from a [120 x 30] row-major
// f32 weight matrix, N-tile `nt` (columns nt*16 .. nt*16+15 of W^T), zero
// padded for K>=30 and gate rows >=120.
// Assumed B layout (wave32): lane l holds column n = l&15; lanes 0-15 hold
// K = {0..7,16..23}, lanes 16-31 hold K = {8..15,24..31}.
__device__ __forceinline__ v16h load_bfrag(const float* __restrict__ W, int nt, int lane) {
    const int n   = lane & 15;
    const int kb  = (lane >> 4) * 8;
    const int row = nt * 16 + n;            // gate index (row of W)
    v16h w;
#pragma unroll
    for (int q = 0; q < 16; ++q) {
        const int k = kb + (q & 7) + (q >> 3) * 16;
        const float v = (row < G4 && k < HH) ? W[row * HH + k] : 0.0f;
        w[q] = (_Float16)v;
    }
    return w;
}

// Load a 16x32 (MxK) f16 A-fragment from an LDS [16][32] f16 buffer.
// ISA A layout: lanes 0-15 -> M=lane, K pairs {0..7,16..23}; lanes 16-31 ->
// M=lane-16, K pairs {8..15,24..31}. Two aligned 16B LDS loads per lane.
__device__ __forceinline__ v16h load_afrag(const _Float16* hbuf, int lane) {
    const int m  = lane & 15;
    const int kb = (lane >> 4) * 8;
    v8h lo = *(const v8h*)(hbuf + m * 32 + kb);
    v8h hi = *(const v8h*)(hbuf + m * 32 + kb + 16);
    return __builtin_shufflevector(lo, hi, 0,1,2,3,4,5,6,7,8,9,10,11,12,13,14,15);
}

__global__ void __launch_bounds__(128)
lstm2_wmma_kernel(const float* __restrict__ x,
                  const float* __restrict__ Wih0, const float* __restrict__ Whh0,
                  const float* __restrict__ bih0, const float* __restrict__ bhh0,
                  const float* __restrict__ Wih1, const float* __restrict__ Whh1,
                  const float* __restrict__ bih1, const float* __restrict__ bhh1,
                  const float* __restrict__ Wout, float* __restrict__ out)
{
    __shared__ __align__(16) _Float16 hbuf0[16 * 32];   // layer0 h (f16, K-padded)
    __shared__ __align__(16) _Float16 hbuf1[16 * 32];   // layer1 h
    __shared__ __align__(16) float    gates[16 * 128];  // one layer's gates at a time
    __shared__ __align__(32) float    xstage[CH * 16];  // x chunk, [t][b] transposed

    const int tid  = threadIdx.x;
    const int lane = tid & 31;
    const int wv   = tid >> 5;          // wave 0..3, owns N-tiles {2wv, 2wv+1}
    const int b0   = blockIdx.x * 16;   // batch tile base
    const int hi8  = (lane >> 4) * 8;   // M offset of this lane's C/D rows

    // h0 = h1 = 0
    for (int i = tid; i < 16 * 32; i += 128) {
        hbuf0[i] = (_Float16)0.f;
        hbuf1[i] = (_Float16)0.f;
    }

    // Per-wave constant fragments & per-lane bias columns (live whole kernel).
    v16h fhh0[2], fhh1[2], fih1[2];
    float bs0[2], wi0[2], bs1[2];
#pragma unroll
    for (int j = 0; j < 2; ++j) {
        const int nt = wv * 2 + j;
        fhh0[j] = load_bfrag(Whh0, nt, lane);
        fhh1[j] = load_bfrag(Whh1, nt, lane);
        fih1[j] = load_bfrag(Wih1, nt, lane);
        const int g = nt * 16 + (lane & 15);
        bs0[j] = (g < G4) ? (bih0[g] + bhh0[g]) : 0.0f;
        wi0[j] = (g < G4) ? Wih0[g] : 0.0f;       // Wih0 is [120 x 1]
        bs1[j] = (g < G4) ? (bih1[g] + bhh1[g]) : 0.0f;
    }

    // Element-wise phase mapping, computed once: thread tid<120 owns column
    // j = tid%30 of batch rows b = (tid/30)*4 + it, it = 0..3.
    const int  ew_q      = tid / HH;
    const int  ew_j      = tid - ew_q * HH;
    const bool ew_active = (tid < G4);

    // Cell states: c[layer][it] for (b = ew_q*4+it, j = ew_j).
    float c0s[4] = {0.f, 0.f, 0.f, 0.f};
    float c1s[4] = {0.f, 0.f, 0.f, 0.f};

    // x chunk prefetch registers: element e = p*128+tid -> (b = e>>5, ts = e&31)
    float xreg[4];
#pragma unroll
    for (int p = 0; p < 4; ++p) {
        const int e = p * 128 + tid;
        xreg[p] = x[(b0 + (e >> 5)) * TT + (e & 31)];
    }

    __syncthreads();

    for (int tc = 0; tc < TT / CH; ++tc) {
        // Commit prefetched x chunk into LDS, transposed to [ts][b].
#pragma unroll
        for (int p = 0; p < 4; ++p) {
            const int e = p * 128 + tid;
            xstage[(e & 31) * 16 + (e >> 5)] = xreg[p];
        }
        __syncthreads();

        // Issue next chunk's global loads now; 32 steps of compute hide them.
        if (tc + 1 < TT / CH) {
#pragma unroll
            for (int p = 0; p < 4; ++p) {
                const int e = p * 128 + tid;
                xreg[p] = x[(b0 + (e >> 5)) * TT + (tc + 1) * CH + (e & 31)];
            }
        }

        for (int ts = 0; ts < CH; ++ts) {
            // ---- layer 0: gates = h0 @ Whh0^T + b + x_t * Wih0 (rank-1 in C)
            {
                v16h a  = load_afrag(hbuf0, lane);
                v8f  xv = *(const v8f*)&xstage[ts * 16 + hi8];  // 8 batch rows
#pragma unroll
                for (int j = 0; j < 2; ++j) {
                    const int nt = wv * 2 + j;
                    v8f c;
#pragma unroll
                    for (int r = 0; r < 8; ++r)
                        c[r] = bs0[j] + xv[r] * wi0[j];
                    v8f d = __builtin_amdgcn_wmma_f32_16x16x32_f16(
                                false, a, false, fhh0[j], (short)0, c, false, false);
#pragma unroll
                    for (int r = 0; r < 8; ++r)
                        gates[(hi8 + r) * 128 + nt * 16 + (lane & 15)] = d[r];
                }
            }
            __syncthreads();

            // ---- layer 0 element-wise (i,f,g,o), update c0, write h0 (f16)
            if (ew_active) {
#pragma unroll
                for (int it = 0; it < 4; ++it) {
                    const int b = ew_q * 4 + it;
                    const float* g = &gates[b * 128 + ew_j];
                    const float cn = sigf(g[HH]) * c0s[it]
                                   + sigf(g[0]) * tanh_(g[2 * HH]);
                    c0s[it] = cn;
                    hbuf0[b * 32 + ew_j] = (_Float16)(sigf(g[3 * HH]) * tanh_(cn));
                }
            }
            __syncthreads();

            // ---- layer 1: gates = y0 @ Wih1^T + h1 @ Whh1^T + b (chained)
            {
                v16h ah = load_afrag(hbuf1, lane);   // h1 from t-1
                v16h ay = load_afrag(hbuf0, lane);   // y0 = new h0 of this step
#pragma unroll
                for (int j = 0; j < 2; ++j) {
                    const int nt = wv * 2 + j;
                    v8f c;
#pragma unroll
                    for (int r = 0; r < 8; ++r) c[r] = bs1[j];
                    v8f d0 = __builtin_amdgcn_wmma_f32_16x16x32_f16(
                                 false, ah, false, fhh1[j], (short)0, c,  false, false);
                    v8f d  = __builtin_amdgcn_wmma_f32_16x16x32_f16(
                                 false, ay, false, fih1[j], (short)0, d0, false, false);
#pragma unroll
                    for (int r = 0; r < 8; ++r)
                        gates[(hi8 + r) * 128 + nt * 16 + (lane & 15)] = d[r];
                }
            }
            __syncthreads();

            // ---- layer 1 element-wise, update c1, write h1
            if (ew_active) {
#pragma unroll
                for (int it = 0; it < 4; ++it) {
                    const int b = ew_q * 4 + it;
                    const float* g = &gates[b * 128 + ew_j];
                    const float cn = sigf(g[HH]) * c1s[it]
                                   + sigf(g[0]) * tanh_(g[2 * HH]);
                    c1s[it] = cn;
                    hbuf1[b * 32 + ew_j] = (_Float16)(sigf(g[3 * HH]) * tanh_(cn));
                }
            }
            __syncthreads();
        }
    }

    // ---- readout: out[b] = sigmoid(relu(h1_T) . Wout) ----
    if (tid < 16) {
        float acc = 0.f;
#pragma unroll
        for (int j = 0; j < HH; ++j) {
            const float h = (float)hbuf1[tid * 32 + j];
            acc += fmaxf(h, 0.f) * Wout[j];
        }
        out[b0 + tid] = sigf(acc);
    }
}

extern "C" void kernel_launch(void* const* d_in, const int* in_sizes, int n_in,
                              void* d_out, int out_size, void* d_ws, size_t ws_size,
                              hipStream_t stream) {
    (void)in_sizes; (void)n_in; (void)out_size; (void)d_ws; (void)ws_size;
    const float* x    = (const float*)d_in[0];
    const float* Wih0 = (const float*)d_in[1];
    const float* Whh0 = (const float*)d_in[2];
    const float* bih0 = (const float*)d_in[3];
    const float* bhh0 = (const float*)d_in[4];
    const float* Wih1 = (const float*)d_in[5];
    const float* Whh1 = (const float*)d_in[6];
    const float* bih1 = (const float*)d_in[7];
    const float* bhh1 = (const float*)d_in[8];
    const float* Wout = (const float*)d_in[9];
    float* out = (float*)d_out;

    lstm2_wmma_kernel<<<dim3(BB / 16), dim3(128), 0, stream>>>(
        x, Wih0, Whh0, bih0, bhh0, Wih1, Whh1, bih1, bhh1, Wout, out);
}
